// SelfAttentionBlock_30030411334228
// MI455X (gfx1250) — matrix operand order repaired and checked by reading, not compiled, not run on hardware
//
#include <hip/hip_runtime.h>
#include <hip/hip_bf16.h>

// ---------- problem constants ----------
#define B_BATCH 4
#define C_DIM   512
#define N_TOK   4096   // 64*64
#define DQK     64     // C/8
#define KBLK    256    // keys staged per block iteration (32 per wave)

typedef __attribute__((ext_vector_type(16))) __bf16 v16bf;
typedef __attribute__((ext_vector_type(8)))  __bf16 v8bf;
typedef __attribute__((ext_vector_type(8)))  float  v8f;
typedef __attribute__((ext_vector_type(4)))  unsigned int u32x4;
typedef __attribute__((ext_vector_type(8)))  int          i32x8;
typedef __attribute__((ext_vector_type(4)))  int          i32x4;

union V16U { v16bf v; v8bf h[2]; };

#if defined(__has_builtin)
#  if __has_builtin(__builtin_amdgcn_tensor_load_to_lds)
#    define HAS_TDM 1
#  endif
#  if __has_builtin(__builtin_amdgcn_s_wait_tensorcnt)
#    define HAS_TENSORCNT 1
#  endif
#endif
#ifndef HAS_TDM
#  define HAS_TDM 0
#endif
#ifndef HAS_TENSORCNT
#  define HAS_TENSORCNT 0
#endif

// =====================================================================
// Kernel 1: Y = relu(W @ X + b)   (per batch)
//   W: (O x C) f32 row-major, X: (C x N) f32, Y bf16.
//   transposed==1 -> store Y as [N][O]  (Q,K: contiguous rows of 64 d-values)
//   transposed==0 -> store Y as [O][N]  (V: contiguous rows over tokens)
// One wave computes one 16(o) x 16(n) D tile with v_wmma_f32_16x16x32_bf16.
// =====================================================================
__global__ __launch_bounds__(32)
void qkv_gemm_kernel(const float* __restrict__ X, const float* __restrict__ W,
                     const float* __restrict__ bias, __bf16* __restrict__ Y,
                     int O, int transposed)
{
    const int lane = threadIdx.x & 31;
    const int hi   = lane >> 4;
    const int l16  = lane & 15;
    const int n0   = blockIdx.x * 16;
    const int o0   = blockIdx.y * 16;
    const int b    = blockIdx.z;

    const float* Xb   = X + (size_t)b * C_DIM * N_TOK;
    const float* wrow = W + (size_t)(o0 + l16) * C_DIM;

    v8f acc = {};
    for (int c0 = 0; c0 < C_DIM; c0 += 32) {
        // A tile (16 o x 32 c): lanes 0-15 hold K 0..7 & 16..23; lanes 16-31 hold K 8..15 & 24..31
        float wa[16];
        *(float4*)&wa[0]  = *(const float4*)(wrow + c0 + 8 * hi);
        *(float4*)&wa[4]  = *(const float4*)(wrow + c0 + 8 * hi + 4);
        *(float4*)&wa[8]  = *(const float4*)(wrow + c0 + 16 + 8 * hi);
        *(float4*)&wa[12] = *(const float4*)(wrow + c0 + 16 + 8 * hi + 4);
        v16bf aW;
        #pragma unroll
        for (int e = 0; e < 16; ++e) aW[e] = (__bf16)wa[e];

        // B tile (32 c x 16 n): lane l16 = column n, lanes 0-15 hold K 0..7 & 16..23; lanes 16-31 hold K 8..15 & 24..31
        v16bf bX;
        #pragma unroll
        for (int kk = 0; kk < 16; ++kk)
            bX[kk] = (__bf16)Xb[(size_t)(c0 + 8 * hi + kk + 8 * (kk >> 3)) * N_TOK + n0 + l16];

        acc = __builtin_amdgcn_wmma_f32_16x16x32_bf16(false, aW, false, bX,
                                                      (short)0, acc, false, false);
    }

    // C/D layout: VGPR r -> row o0 + r + 8*hi, column n0 + l16
    if (transposed) {
        v8bf ov;
        #pragma unroll
        for (int r = 0; r < 8; ++r) {
            float val = acc[r] + bias[o0 + 8 * hi + r];
            ov[r] = (__bf16)fmaxf(val, 0.0f);
        }
        *(v8bf*)(Y + (size_t)b * N_TOK * O + (size_t)(n0 + l16) * O + o0 + 8 * hi) = ov;
    } else {
        #pragma unroll
        for (int r = 0; r < 8; ++r) {
            float val = acc[r] + bias[o0 + 8 * hi + r];
            Y[(size_t)b * O * N_TOK + (size_t)(o0 + 8 * hi + r) * N_TOK + n0 + l16] =
                (__bf16)fmaxf(val, 0.0f);
        }
    }
}

// =====================================================================
// TDM: 2D tensor descriptor load of a (KBLK x DQK) bf16 tile into LDS.
// D# per CDNA5 ISA 8.3/8.4: group0 = {count, lds_addr, global_addr, type=2},
// group1 = {mask=0, data_size=2B, tensor_dim0=64, tensor_dim1=N, tile=64x256,
//           dim0_stride=64}. Groups 2/3 zero (<=2D tensor).
// =====================================================================
__device__ __forceinline__ void tdm_load_k_tile(unsigned int lds_addr, const __bf16* gptr)
{
#if HAS_TDM
    unsigned long long ga = (unsigned long long)(size_t)gptr;
    u32x4 g0;
    g0[0] = 1u;                                   // count=1, no gather
    g0[1] = lds_addr;                             // LDS byte address
    g0[2] = (unsigned int)ga;                     // global_addr[31:0]
    g0[3] = (unsigned int)(ga >> 32) | (2u << 30);// global_addr[56:32] | type=2
    i32x8 g1;
    g1[0] = 0x00010000;                           // workgroup_mask=0, data_size=1 (2 bytes)
    g1[1] = (DQK << 16);                          // tensor_dim0[15:0]=64 (bits 63:48)
    g1[2] = (N_TOK << 16);                        // tensor_dim1[15:0]=4096 (bits 95:80)
    g1[3] = (DQK << 16);                          // tile_dim0=64 (bits 127:112)
    g1[4] = KBLK;                                 // tile_dim1=256
    g1[5] = DQK;                                  // tensor_dim0_stride=64
    g1[6] = 0;
    g1[7] = 0;
    i32x4 gz = {0, 0, 0, 0};
#if defined(__clang_major__) && (__clang_major__ >= 23)
    i32x8 gz8 = {0, 0, 0, 0, 0, 0, 0, 0};
    __builtin_amdgcn_tensor_load_to_lds(g0, g1, gz, gz, gz8, 0);
#else
    __builtin_amdgcn_tensor_load_to_lds(g0, g1, gz, gz, 0);
#endif
#if HAS_TENSORCNT
    __builtin_amdgcn_s_wait_tensorcnt(0);
#else
    asm volatile("s_wait_tensorcnt 0x0" ::: "memory");
#endif
#else
    (void)lds_addr; (void)gptr;
#endif
}

// =====================================================================
// Kernel 2: cooperative flash-attention + residual.
//   q,k : [B][N][64]  bf16 ; v : [B][C][N] bf16
//   out = gamma * (softmax(QK^T) V^T) + x, f32 [B][C][N]
// Block = 16 queries x 512 channels, 8 waves.
// Per 256-key step: TDM stages K into LDS; wave w computes S for its own
// 32-key chunk (no redundancy); softmax stats merged through LDS partials;
// P tiles shared in LDS; every wave runs 8 PV WMMAs x 4 channel tiles.
// =====================================================================
struct SMem {
    __bf16 kbuf[KBLK * DQK];   // 32 KB staged K tile
    __bf16 pbuf[16 * KBLK];    //  8 KB P (16 queries x 256 keys)
    float  maxp[8 * 16];       // per-wave partial row max
    float  sump[8 * 16];       // per-wave partial row sum
};

__global__ __launch_bounds__(256)
void attn_kernel(const __bf16* __restrict__ qt, const __bf16* __restrict__ kt,
                 const __bf16* __restrict__ vt, const float* __restrict__ x,
                 const float* __restrict__ gamma_p, float* __restrict__ out)
{
    __shared__ SMem sm;

    const int lane = threadIdx.x & 31;
    const int wave = threadIdx.x >> 5;
    const int hi   = lane >> 4;
    const int l16  = lane & 15;
    const int i0   = blockIdx.x * 16;     // query tile
    const int b    = blockIdx.y;
    const int cb   = wave * 64;           // this wave's 4 channel tiles
    const int jc   = wave * 32;           // this wave's key chunk inside kbuf

    const __bf16* q = qt + (size_t)b * N_TOK * DQK;
    const __bf16* k = kt + (size_t)b * N_TOK * DQK;
    const __bf16* v = vt + (size_t)b * C_DIM * N_TOK;

    // Q tile in A layout for d=0..31 and d=32..63, kept in registers
    V16U aQ0, aQ1;
    {
        const __bf16* qrow = q + (size_t)(i0 + l16) * DQK;
        aQ0.h[0] = *(const v8bf*)(qrow + 16 * hi);
        aQ0.h[1] = *(const v8bf*)(qrow + 8 + 16 * hi);
        aQ1.h[0] = *(const v8bf*)(qrow + 32 + 16 * hi);
        aQ1.h[1] = *(const v8bf*)(qrow + 40 + 16 * hi);
    }

    float m_r[8], l_r[8];
    v8f acc[4] = {};
    #pragma unroll
    for (int r = 0; r < 8; ++r) { m_r[r] = -3.0e38f; l_r[r] = 0.0f; }

    const unsigned int kbuf_lds = (unsigned int)(size_t)(void*)sm.kbuf;

    for (int j0 = 0; j0 < N_TOK; j0 += KBLK) {
        // ---- (A) stage 256 K rows into LDS ----
#if HAS_TDM
        if (threadIdx.x == 0)
            tdm_load_k_tile(kbuf_lds, k + (size_t)j0 * DQK);  // EXEC ignored by TDM
#else
        {   // cooperative fallback copy: 256 rows x 128B
            const uint4* src = (const uint4*)(k + (size_t)j0 * DQK);
            uint4* dst = (uint4*)sm.kbuf;
            for (int idx = threadIdx.x; idx < KBLK * DQK / 8; idx += 256)
                dst[idx] = src[idx];
        }
#endif
        __syncthreads();

        if (j0 + KBLK < N_TOK)
            __builtin_prefetch(v + (size_t)(cb + l16) * N_TOK + j0 + KBLK, 0, 0);

        // ---- (B) S for this wave's 32-key chunk, from LDS K ----
        v16bf bKl0 = *(const v16bf*)&sm.kbuf[(jc + l16) * DQK + 16 * hi];
        v16bf bKl1 = *(const v16bf*)&sm.kbuf[(jc + l16) * DQK + 32 + 16 * hi];
        v16bf bKr0 = *(const v16bf*)&sm.kbuf[(jc + 16 + l16) * DQK + 16 * hi];
        v16bf bKr1 = *(const v16bf*)&sm.kbuf[(jc + 16 + l16) * DQK + 32 + 16 * hi];

        v8f Sl = {}, Sr = {};
        Sl = __builtin_amdgcn_wmma_f32_16x16x32_bf16(false, aQ0.v, false, bKl0, (short)0, Sl, false, false);
        Sl = __builtin_amdgcn_wmma_f32_16x16x32_bf16(false, aQ1.v, false, bKl1, (short)0, Sl, false, false);
        Sr = __builtin_amdgcn_wmma_f32_16x16x32_bf16(false, aQ0.v, false, bKr0, (short)0, Sr, false, false);
        Sr = __builtin_amdgcn_wmma_f32_16x16x32_bf16(false, aQ1.v, false, bKr1, (short)0, Sr, false, false);

        // partial row max (VGPR r holds row r for lanes 0-15, row r+8 for lanes 16-31)
        float mx8[8];
        #pragma unroll
        for (int r = 0; r < 8; ++r) {
            float mx = fmaxf(Sl[r], Sr[r]);
            mx = fmaxf(mx, __shfl_xor(mx, 1, 32));
            mx = fmaxf(mx, __shfl_xor(mx, 2, 32));
            mx = fmaxf(mx, __shfl_xor(mx, 4, 32));
            mx = fmaxf(mx, __shfl_xor(mx, 8, 32));
            mx8[r] = mx;
        }
        if (l16 == 0) {
            #pragma unroll
            for (int r = 0; r < 8; ++r) sm.maxp[wave * 16 + r + 8 * hi] = mx8[r];
        }
        __syncthreads();

        // ---- (C) merge row max, exponentiate own chunk, share P + partial sums ----
        float sc[8], pl[8], pr[8], ps[8];
        #pragma unroll
        for (int r = 0; r < 8; ++r) {
            float mn = m_r[r];
            #pragma unroll
            for (int w = 0; w < 8; ++w) mn = fmaxf(mn, sm.maxp[w * 16 + r + 8 * hi]);
            sc[r] = __expf(m_r[r] - mn);
            m_r[r] = mn;
            pl[r] = __expf(Sl[r] - mn);
            pr[r] = __expf(Sr[r] - mn);
            float s = pl[r] + pr[r];
            s += __shfl_xor(s, 1, 32);
            s += __shfl_xor(s, 2, 32);
            s += __shfl_xor(s, 4, 32);
            s += __shfl_xor(s, 8, 32);
            ps[r] = s;
        }
        if (l16 == 0) {
            #pragma unroll
            for (int r = 0; r < 8; ++r) sm.sump[wave * 16 + r + 8 * hi] = ps[r];
        }
        #pragma unroll
        for (int r = 0; r < 8; ++r) {
            int row = r + 8 * hi;
            sm.pbuf[row * KBLK + jc + l16]      = (__bf16)pl[r];
            sm.pbuf[row * KBLK + jc + 16 + l16] = (__bf16)pr[r];
        }
        __syncthreads();

        // ---- (D) update l, rescale acc, PV over all 8 shared P tiles ----
        #pragma unroll
        for (int r = 0; r < 8; ++r) {
            float ssum = 0.0f;
            #pragma unroll
            for (int w = 0; w < 8; ++w) ssum += sm.sump[w * 16 + r + 8 * hi];
            l_r[r] = l_r[r] * sc[r] + ssum;
            #pragma unroll
            for (int t = 0; t < 4; ++t) acc[t][r] *= sc[r];
        }
        for (int p = 0; p < 8; ++p) {
            V16U aP;   // A layout: lane row l16; K cols p*32 + {0..15} (+16 for upper half)
            aP.h[0] = *(const v8bf*)&sm.pbuf[l16 * KBLK + p * 32 + 16 * hi];
            aP.h[1] = *(const v8bf*)&sm.pbuf[l16 * KBLK + p * 32 + 8 + 16 * hi];
            #pragma unroll
            for (int t = 0; t < 4; ++t) {
                const __bf16* vrow = v + (size_t)(cb + t * 16 + l16) * N_TOK + j0 + p * 32 + 16 * hi;
                v16bf bV = *(const v16bf*)vrow;
                acc[t] = __builtin_amdgcn_wmma_f32_16x16x32_bf16(false, aP.v, false, bV,
                                                                 (short)0, acc[t], false, false);
            }
        }
        __syncthreads();
    }

    // ---- epilogue: out[b][c][i] = gamma * acc/l + x[b][c][i] ----
    const float g = gamma_p[0];
    const float* xb = x   + (size_t)b * C_DIM * N_TOK;
    float*       ob = out + (size_t)b * C_DIM * N_TOK;
    float inv[8];
    #pragma unroll
    for (int r = 0; r < 8; ++r) inv[r] = 1.0f / l_r[r];

    #pragma unroll
    for (int t = 0; t < 4; ++t) {
        int c = cb + t * 16 + l16;
        size_t base = (size_t)c * N_TOK + i0 + 8 * hi;   // 8 consecutive i per lane
        #pragma unroll
        for (int h = 0; h < 2; ++h) {
            float4 xv = *(const float4*)(xb + base + 4 * h);
            float4 ov;
            ov.x = g * acc[t][4 * h + 0] * inv[4 * h + 0] + xv.x;
            ov.y = g * acc[t][4 * h + 1] * inv[4 * h + 1] + xv.y;
            ov.z = g * acc[t][4 * h + 2] * inv[4 * h + 2] + xv.z;
            ov.w = g * acc[t][4 * h + 3] * inv[4 * h + 3] + xv.w;
            *(float4*)(ob + base + 4 * h) = ov;
        }
    }
}

// =====================================================================
// launch
// =====================================================================
extern "C" void kernel_launch(void* const* d_in, const int* in_sizes, int n_in,
                              void* d_out, int out_size, void* d_ws, size_t ws_size,
                              hipStream_t stream)
{
    const float* x     = (const float*)d_in[0];
    const float* wq    = (const float*)d_in[1];
    const float* bq    = (const float*)d_in[2];
    const float* wk    = (const float*)d_in[3];
    const float* bk    = (const float*)d_in[4];
    const float* wv    = (const float*)d_in[5];
    const float* bv    = (const float*)d_in[6];
    const float* gamma = (const float*)d_in[7];
    float* out = (float*)d_out;

    // workspace (bf16): q_t [B][N][64], k_t [B][N][64], v [B][C][N]
    __bf16* qt = (__bf16*)d_ws;
    __bf16* kt = qt + (size_t)B_BATCH * N_TOK * DQK;
    __bf16* vt = kt + (size_t)B_BATCH * N_TOK * DQK;

    dim3 gq(N_TOK / 16, DQK / 16, B_BATCH);    // 256 x 4 x 4
    dim3 gv(N_TOK / 16, C_DIM / 16, B_BATCH);  // 256 x 32 x 4

    qkv_gemm_kernel<<<gq, 32, 0, stream>>>(x, wq, bq, qt, DQK, 1);
    qkv_gemm_kernel<<<gq, 32, 0, stream>>>(x, wk, bk, kt, DQK, 1);
    qkv_gemm_kernel<<<gv, 32, 0, stream>>>(x, wv, bv, vt, C_DIM, 0);

    dim3 ga(N_TOK / 16, B_BATCH);              // 256 x 4 blocks, 8 waves each
    attn_kernel<<<ga, 256, 0, stream>>>(qt, kt, vt, x, gamma, out);
}
